// MDyGraphConv2d_57114475102440
// MI455X (gfx1250) — matrix operand order repaired and checked by reading, not compiled, not run on hardware
//
#include <hip/hip_runtime.h>
#include <hip/hip_bf16.h>

// ---------------------------------------------------------------------------
// MDyGraphConv2d for MI455X (gfx1250), wave32 + v_wmma_f32_16x16x32_bf16
// B=2, C=128, NX=NY=4096, N=8192/batch, K = 9 inner + 3 cross, 2 conv layers
// ---------------------------------------------------------------------------

typedef __attribute__((ext_vector_type(16))) __bf16         v16bf;
typedef __attribute__((ext_vector_type(8)))  float          v8f;
typedef __attribute__((ext_vector_type(4)))  float          v4f;
typedef unsigned short                                      ushort_t;
typedef __attribute__((ext_vector_type(8)))  unsigned short ushort8;

union BF16x16 { v16bf v; ushort_t u[16]; };

__device__ __forceinline__ ushort_t f2bf(float f) {
  union { float f; unsigned int u; } x; x.f = f;
  unsigned int r = x.u + 0x7FFFu + ((x.u >> 16) & 1u);   // round-to-nearest-even
  return (ushort_t)(r >> 16);
}

// ---- WMMA operand loaders (bf16, per CDNA5 ISA VGPR layouts) ----
// A-matrix 16x32 (MxK): lane holds row M=lane%16; halves 0..7 at K = kc + (hi?8:0)..,
// halves 8..15 at K = kc+16 + (hi?8:0)..   (hi = lane>=16)
__device__ __forceinline__ v16bf load_amat(const ushort_t* __restrict__ row, int kc, int lane) {
  const ushort_t* p = row + kc + ((lane & 16) ? 8 : 0);
  ushort8 lo = *(const ushort8*)(p);
  ushort8 hi = *(const ushort8*)(p + 16);
  BF16x16 r;
#pragma unroll
  for (int i = 0; i < 8; ++i) { r.u[i] = lo[i]; r.u[8 + i] = hi[i]; }
  return r.v;
}
// B-matrix 32x16 (KxN): lane holds column N=lane%16; halves 0..7 at
// K = kc + (hi?8:0).., halves 8..15 at K = kc+16 + (hi?8:0)..
__device__ __forceinline__ v16bf load_bmat(const ushort_t* __restrict__ row, int kc, int lane) {
  const ushort_t* p = row + kc + ((lane & 16) ? 8 : 0);
  ushort8 lo = *(const ushort8*)(p);
  ushort8 hi = *(const ushort8*)(p + 16);
  BF16x16 r;
#pragma unroll
  for (int i = 0; i < 8; ++i) { r.u[i] = lo[i]; r.u[8 + i] = hi[i]; }
  return r.v;
}

// ---- branchless sorted top-k insertion (ascending; bv[0] = best) ----------
template <int K>
__device__ __forceinline__ void topk_insert(float (&bv)[K], int (&bi)[K], float v, int id) {
#pragma unroll
  for (int j = 0; j < K; ++j) {
    bool c = v < bv[j];
    float tv = bv[j]; int ti = bi[j];
    bv[j] = c ? v : tv;  bi[j] = c ? id : ti;
    v     = c ? tv : v;  id    = c ? ti : id;
  }
}

// ---------------------------------------------------------------------------
// Kernel 1: pack x/y ([b][c][n] f32) -> feat f32 [row][c], featbf bf16, norms
// ---------------------------------------------------------------------------
__global__ void pack_kernel(const float* __restrict__ x, const float* __restrict__ y,
                            ushort_t* __restrict__ featbf, float* __restrict__ feat,
                            float* __restrict__ norms) {
  const int row = blockIdx.x;            // b*8192 + nn
  const int c   = threadIdx.x;           // 0..127
  const int b   = row >> 13;
  const int nn  = row & 8191;
  float v;
  if (nn < 4096) v = x[((size_t)b * 128 + c) * 4096 + nn];
  else           v = y[((size_t)b * 128 + c) * 4096 + (nn - 4096)];
  feat[(size_t)row * 128 + c]   = v;
  featbf[(size_t)row * 128 + c] = f2bf(v);
  float s = v * v;
#pragma unroll
  for (int d = 16; d > 0; d >>= 1) s += __shfl_xor(s, d, 32);
  __shared__ float sred[4];
  if ((threadIdx.x & 31) == 0) sred[threadIdx.x >> 5] = s;
  __syncthreads();
  if (threadIdx.x == 0) norms[row] = sred[0] + sred[1] + sred[2] + sred[3];
}

__global__ void packw_kernel(const float* __restrict__ W, ushort_t* __restrict__ Wbf, int n) {
  int i = blockIdx.x * blockDim.x + threadIdx.x;
  if (i < n) Wbf[i] = f2bf(W[i]);
}

// ---------------------------------------------------------------------------
// Kernel 2: fused distance-GEMM + top-k KNN.
// Candidates on M, queries on N -> each lane streams candidates for 1 query.
// KSEL==9: blockIdx.y selects x-x / y-y.  KSEL==3: x->y / y->x cross edges.
// ---------------------------------------------------------------------------
template <int KSEL>
__global__ void knn_kernel(const ushort_t* __restrict__ featbf,
                           const float* __restrict__ norms,
                           int* __restrict__ nbr) {
  const int lane = threadIdx.x & 31;
  const int wave = threadIdx.x >> 5;     // 0..3
  const int b    = blockIdx.z;
  int qSegY, cSegY, slot;
  if (KSEL == 9) { qSegY = blockIdx.y; cSegY = blockIdx.y;           slot = 0; }
  else           { qSegY = blockIdx.y; cSegY = 1 - (int)blockIdx.y;  slot = 9; }
  const int base = b * 8192;
  const ushort_t* qmat = featbf + (size_t)(base + qSegY * 4096) * 128;
  const ushort_t* cmat = featbf + (size_t)(base + cSegY * 4096) * 128;
  const float*    cn   = norms + base + cSegY * 4096;
  const int idxOff = cSegY ? 4096 : 0;

  const int qBase = ((int)blockIdx.x * 4 + wave) * 16;
  const ushort_t* qrow = qmat + (size_t)(qBase + (lane & 15)) * 128;
  v16bf qop0 = load_bmat(qrow, 0,  lane);
  v16bf qop1 = load_bmat(qrow, 32, lane);
  v16bf qop2 = load_bmat(qrow, 64, lane);
  v16bf qop3 = load_bmat(qrow, 96, lane);

  float bv[KSEL]; int bi[KSEL];
#pragma unroll
  for (int j = 0; j < KSEL; ++j) { bv[j] = 3.0e38f; bi[j] = 0; }

  for (int cb = 0; cb < 4096; cb += 16) {
    int pf = (cb + 64 <= 4080) ? cb + 64 : cb;
    __builtin_prefetch(cmat + (size_t)(pf + (lane & 15)) * 128, 0, 1);

    const ushort_t* crow = cmat + (size_t)(cb + (lane & 15)) * 128;
    v8f acc = {0.f, 0.f, 0.f, 0.f, 0.f, 0.f, 0.f, 0.f};
    acc = __builtin_amdgcn_wmma_f32_16x16x32_bf16(false, load_amat(crow, 0,  lane),
                                                  false, qop0, (short)0, acc, false, false);
    acc = __builtin_amdgcn_wmma_f32_16x16x32_bf16(false, load_amat(crow, 32, lane),
                                                  false, qop1, (short)0, acc, false, false);
    acc = __builtin_amdgcn_wmma_f32_16x16x32_bf16(false, load_amat(crow, 64, lane),
                                                  false, qop2, (short)0, acc, false, false);
    acc = __builtin_amdgcn_wmma_f32_16x16x32_bf16(false, load_amat(crow, 96, lane),
                                                  false, qop3, (short)0, acc, false, false);

    // score = |b_j|^2 - 2 a.b  (the |a|^2 term is ranking-invariant)
    const float* np = cn + cb + ((lane & 16) ? 8 : 0);
    v4f n0 = *(const v4f*)np;
    v4f n1 = *(const v4f*)(np + 4);
    const int mBase = cb + ((lane & 16) ? 8 : 0);
#pragma unroll
    for (int r = 0; r < 8; ++r) {
      float nv = (r < 4) ? n0[r] : n1[r - 4];
      float s  = nv - 2.0f * acc[r];
      if (s < bv[KSEL - 1]) topk_insert<KSEL>(bv, bi, s, mBase + r);
    }
  }

  // lanes l and l+16 hold disjoint candidate halves for the same query: merge
#pragma unroll
  for (int j = 0; j < KSEL; ++j) {
    float pv = __shfl_xor(bv[j], 16, 32);
    int   pi = __shfl_xor(bi[j], 16, 32);
    topk_insert<KSEL>(bv, bi, pv, pi);
  }
  if (lane < 16) {
    const int q   = qBase + lane;
    const int row = base + qSegY * 4096 + q;
    int* dst = nbr + (size_t)row * 12 + slot;
#pragma unroll
    for (int j = 0; j < KSEL; ++j) dst[j] = bi[j] + idxOff;
  }
}

// ---------------------------------------------------------------------------
// Kernel 3: max-relative aggregation -> h = [feat, max_j(feat[nbr]-feat)] bf16
// ---------------------------------------------------------------------------
__global__ void aggregate_kernel(const float* __restrict__ feat, const int* __restrict__ nbr,
                                 ushort_t* __restrict__ hbuf) {
  const int row = blockIdx.x;            // b*8192 + nn
  const int c   = threadIdx.x;           // 0..127
  const int b   = row >> 13;
  const float self = feat[(size_t)row * 128 + c];
  float rel = -3.0e38f;
  const int* nb = nbr + (size_t)row * 12;
#pragma unroll
  for (int j = 0; j < 12; ++j) {
    int idx = nb[j];                     // 0..8191 within this batch
    float v = feat[(size_t)(b * 8192 + idx) * 128 + c];
    float d = v - self;
    rel = d > rel ? d : rel;
  }
  hbuf[(size_t)row * 256 + c]       = f2bf(self);
  hbuf[(size_t)row * 256 + 128 + c] = f2bf(rel);
}

__global__ void zero_stats_kernel(float* __restrict__ s) {
  if (threadIdx.x < 256) s[threadIdx.x] = 0.0f;
}

// ---------------------------------------------------------------------------
// Kernel 4: 1x1 conv GEMM (16384 x 128 x 256, bf16 WMMA) + bias + BN partials
// ---------------------------------------------------------------------------
__global__ void conv_gemm_kernel(const ushort_t* __restrict__ hbuf,
                                 const ushort_t* __restrict__ Wbf,
                                 const float* __restrict__ bias,
                                 float* __restrict__ otmp,
                                 float* __restrict__ stats) {
  const int lane  = threadIdx.x & 31;
  const int wave  = threadIdx.x >> 5;                 // 0..7
  const int tile  = (int)blockIdx.x * 8 + wave;       // 0..8191
  const int mtile = tile >> 3;
  const int ntile = tile & 7;
  const ushort_t* arow = hbuf + (size_t)(mtile * 16 + (lane & 15)) * 256;
  const ushort_t* brow = Wbf  + (size_t)(ntile * 16 + (lane & 15)) * 256;
  v8f acc = {0.f, 0.f, 0.f, 0.f, 0.f, 0.f, 0.f, 0.f};
#pragma unroll
  for (int kc = 0; kc < 256; kc += 32)
    acc = __builtin_amdgcn_wmma_f32_16x16x32_bf16(false, load_amat(arow, kc, lane),
                                                  false, load_bmat(brow, kc, lane),
                                                  (short)0, acc, false, false);
  const int cout = ntile * 16 + (lane & 15);
  const float bvl = bias[cout];
  const int mBase = mtile * 16 + ((lane & 16) ? 8 : 0);
  float ps = 0.f, pss = 0.f;
#pragma unroll
  for (int r = 0; r < 8; ++r) {
    float o = acc[r] + bvl;
    otmp[(size_t)(mBase + r) * 128 + cout] = o;
    ps += o; pss += o * o;
  }
  ps  += __shfl_xor(ps, 16, 32);
  pss += __shfl_xor(pss, 16, 32);
  if (lane < 16) {
    atomicAdd(&stats[cout], ps);
    atomicAdd(&stats[128 + cout], pss);
  }
}

// ---------------------------------------------------------------------------
// Kernel 5: batchnorm (train) + tanh-GELU + residual into feat
// ---------------------------------------------------------------------------
__global__ void bnact_kernel(const float* __restrict__ otmp, const float* __restrict__ stats,
                             const float* __restrict__ gamma, const float* __restrict__ beta,
                             float* __restrict__ feat) {
  int i = blockIdx.x * blockDim.x + threadIdx.x;
  if (i >= 16384 * 128) return;
  const int c = i & 127;
  const float invN = 1.0f / 16384.0f;
  float mean = stats[c] * invN;
  float var  = stats[128 + c] * invN - mean * mean;
  float xh = (otmp[i] - mean) * rsqrtf(var + 1e-5f) * gamma[c] + beta[c];
  float g  = 0.5f * xh * (1.0f + tanhf(0.7978845608028654f * (xh + 0.044715f * xh * xh * xh)));
  feat[i] += g;   // residual
}

// ---------------------------------------------------------------------------
// Kernel 6: transpose feat [row][c] -> outputs [B][C][NX], [B][C][NY]
// ---------------------------------------------------------------------------
__global__ void output_kernel(const float* __restrict__ feat, float* __restrict__ out) {
  int i = blockIdx.x * blockDim.x + threadIdx.x;
  if (i >= 16384 * 128) return;
  const int row = i >> 7;
  const int c   = i & 127;
  const int b   = row >> 13;
  const int nn  = row & 8191;
  const float v = feat[i];
  if (nn < 4096) out[((size_t)b * 128 + c) * 4096 + nn] = v;
  else out[1048576 + ((size_t)b * 128 + c) * 4096 + (nn - 4096)] = v;
}

// ---------------------------------------------------------------------------
extern "C" void kernel_launch(void* const* d_in, const int* in_sizes, int n_in,
                              void* d_out, int out_size, void* d_ws, size_t ws_size,
                              hipStream_t stream) {
  (void)in_sizes; (void)n_in; (void)out_size; (void)ws_size;
  const float* x     = (const float*)d_in[0];
  const float* y     = (const float*)d_in[1];
  const float* W     = (const float*)d_in[2];
  const float* bias  = (const float*)d_in[3];
  const float* gamma = (const float*)d_in[4];
  const float* beta  = (const float*)d_in[5];
  float* out = (float*)d_out;

  char* ws = (char*)d_ws;
  size_t off = 0;
  auto alloc = [&](size_t bytes) -> void* {
    void* p = ws + off;
    off = (off + bytes + 255) & ~(size_t)255;
    return p;
  };
  ushort_t* featbf = (ushort_t*)alloc(16384ull * 128 * 2);  // bf16 features
  float*    norms  = (float*)   alloc(16384ull * 4);        // |f|^2 per node
  int*      nbr    = (int*)     alloc(16384ull * 12 * 4);   // 12 neighbors/node
  float*    feat   = (float*)   alloc(16384ull * 128 * 4);  // residual stream (f32)
  ushort_t* hbuf   = (ushort_t*)alloc(16384ull * 256 * 2);  // [feat, rel] bf16
  float*    otmp   = (float*)   alloc(16384ull * 128 * 4);  // conv output (f32)
  ushort_t* Wbf    = (ushort_t*)alloc(2ull * 128 * 256 * 2);// bf16 weights
  float*    stats  = (float*)   alloc(256 * 4);             // [sum(128), sumsq(128)]

  // 1) pack features -> f32 + bf16 + norms
  pack_kernel<<<16384, 128, 0, stream>>>(x, y, featbf, feat, norms);
  packw_kernel<<<(2 * 128 * 256 + 255) / 256, 256, 0, stream>>>(W, Wbf, 2 * 128 * 256);

  // 2) fused WMMA distance + top-k (inner k=9, cross k=3)
  dim3 gknn(64, 2, 2);   // 64 query tiles x {seg pair} x {batch}
  knn_kernel<9><<<gknn, 128, 0, stream>>>(featbf, norms, nbr);
  knn_kernel<3><<<gknn, 128, 0, stream>>>(featbf, norms, nbr);

  // 3) two MRConv layers
  for (int l = 0; l < 2; ++l) {
    aggregate_kernel<<<16384, 128, 0, stream>>>(feat, nbr, hbuf);
    zero_stats_kernel<<<1, 256, 0, stream>>>(stats);
    conv_gemm_kernel<<<1024, 256, 0, stream>>>(hbuf, Wbf + (size_t)l * 128 * 256,
                                               bias + l * 128, otmp, stats);
    bnact_kernel<<<(16384 * 128 + 255) / 256, 256, 0, stream>>>(
        otmp, stats, gamma + l * 128, beta + l * 128, feat);
  }

  // 4) transpose to output layout
  output_kernel<<<(16384 * 128 + 255) / 256, 256, 0, stream>>>(feat, out);
}
